// NeuralIntraAttentionModel_43611097923988
// MI455X (gfx1250) — compile-verified, matrix operand order checked by reading, not run on hardware
//
#include <hip/hip_runtime.h>
#include <hip/hip_bf16.h>
#include <math.h>

// ---------------------------------------------------------------------------
// CDNA5 WMMA types & helpers
// ---------------------------------------------------------------------------
typedef __attribute__((ext_vector_type(16))) __bf16 v16bf;
typedef __attribute__((ext_vector_type(8)))  float  v8f;

union Frag { v16bf v; unsigned u[8]; };

#define DEVINL static __device__ __forceinline__

DEVINL unsigned short f2bf(float f) {
    union { float f; unsigned u; } x; x.f = f;
    unsigned r = (x.u + 0x7FFFu + ((x.u >> 16) & 1u)) >> 16;
    return (unsigned short)r;
}
DEVINL float bf2f(unsigned short h) {
    union { unsigned u; float f; } x; x.u = ((unsigned)h) << 16;
    return x.f;
}
DEVINL float sigm(float x) { return 1.0f / (1.0f + __expf(-x)); }

// load one bf16 A/B fragment (row-major, K-contiguous per lane) as 8 dwords.
// layout per ISA 7.12.2: lane<16 -> k-half 0, lane>=16 -> k-half +8;
// VGPR p<4 holds k = 2p, 2p+1 ; p>=4 holds k = 16+2(p-4), ...
DEVINL void load_frag(const unsigned short* __restrict__ p, Frag& f) {
#pragma unroll
    for (int i = 0; i < 4; ++i) {
        f.u[i]     = *(const unsigned*)(p + 2 * i);
        f.u[4 + i] = *(const unsigned*)(p + 16 + 2 * i);
    }
}

DEVINL v8f wmma_bf16(const Frag& a, const Frag& b, v8f c) {
    return __builtin_amdgcn_wmma_f32_16x16x32_bf16(false, a.v, false, b.v,
                                                   (short)0, c, false, false);
}

// segmented block reduction: 1024 threads = 8 segments x 128 threads (4 waves)
// result for segment b lands in out8[b].
DEVINL void segreduce(float v, int domax, float* wred, float* out8, int tid) {
    int lane = tid & 31, wave = tid >> 5;
#pragma unroll
    for (int off = 16; off >= 1; off >>= 1) {
        float o = __shfl_xor(v, off);
        v = domax ? fmaxf(v, o) : (v + o);
    }
    if (lane == 0) wred[wave] = v;
    __syncthreads();
    if (tid < 8) {
        float r = wred[tid * 4];
#pragma unroll
        for (int j = 1; j < 4; ++j) {
            float o = wred[tid * 4 + j];
            r = domax ? fmaxf(r, o) : (r + o);
        }
        out8[tid] = r;
    }
    __syncthreads();
}

// ---------------------------------------------------------------------------
// small utility kernels
// ---------------------------------------------------------------------------
__global__ void k_zero_u32(unsigned* p, int n) {
    int i = blockIdx.x * blockDim.x + threadIdx.x;
    if (i < n) p[i] = 0u;
}

__global__ void k_cvt_bf16(const float* __restrict__ in,
                           unsigned short* __restrict__ out, int n) {
    int i = blockIdx.x * blockDim.x + threadIdx.x;
    if (i < n) out[i] = f2bf(in[i]);
}

// in: R x C row-major (f32)  ->  out: C x R (bf16)
__global__ void k_transpose_bf16(const float* __restrict__ in,
                                 unsigned short* __restrict__ out, int R, int C) {
    int i = blockIdx.x * blockDim.x + threadIdx.x;
    if (i < R * C) {
        int r = i / C, c = i % C;
        out[(size_t)c * R + r] = f2bf(in[i]);
    }
}

__global__ void k_gather_enc(const int* __restrict__ ids,
                             const float* __restrict__ emb,
                             unsigned short* __restrict__ out,
                             int BS, int E, int V) {
    int i = blockIdx.x * blockDim.x + threadIdx.x;
    if (i < BS * E) {
        int p = i / E, e = i % E;
        int id = ids[p]; if (id >= V) id = 1;  // UNK
        out[i] = f2bf(emb[(size_t)id * E + e]);
    }
}

// teacher forcing: dec input at t is START for t==0, else mapped labels[:,t-1]
__global__ void k_gather_dec(const int* __restrict__ labels,
                             const float* __restrict__ emb,
                             unsigned short* __restrict__ out,
                             int B, int T, int E, int V) {
    int i = blockIdx.x * blockDim.x + threadIdx.x;
    if (i < B * T * E) {
        int e = i % E, bt = i / E;
        int t = bt % T, b = bt / T;
        int tok = (t == 0) ? 2 : labels[b * T + t - 1];   // START=2
        if (tok >= V) tok = 1;                            // UNK
        out[i] = f2bf(emb[(size_t)tok * E + e]);
    }
}

// ---------------------------------------------------------------------------
// generic bf16 WMMA GEMM : C[M,N] = A[M,K] * W[N,K]^T (+bias) (tanh?)
// one wave per 16x16 tile. M,N multiples of 16; K multiple of 32.
// ---------------------------------------------------------------------------
__global__ __launch_bounds__(256) void k_wmma_gemm(
    const unsigned short* __restrict__ A, const unsigned short* __restrict__ W,
    float* __restrict__ Cf, unsigned short* __restrict__ Cbf,
    const float* __restrict__ bias, int Mt, int Nt, int K, int doTanh)
{
    int wid = blockIdx.x * 8 + (threadIdx.x >> 5);
    int tiles = Mt * Nt;
    if (wid >= tiles) return;
    int mt = wid / Nt, nt = wid % Nt;
    int lane = threadIdx.x & 31;
    int l15 = lane & 15;
    int kh  = (lane >> 4) << 3;
    int m0 = mt << 4, n0 = nt << 4;
    int ks = K >> 5;
    const unsigned short* Arow = A + (size_t)(m0 + l15) * K + kh;
    const unsigned short* Wrow = W + (size_t)(n0 + l15) * K + kh;
    v8f acc = {0, 0, 0, 0, 0, 0, 0, 0};
    for (int kk = 0; kk < ks; ++kk) {
        Frag a, b;
        load_frag(Arow + kk * 32, a);
        load_frag(Wrow + kk * 32, b);
        if (kk + 1 < ks) __builtin_prefetch((const void*)(Wrow + kk * 32 + 32), 0, 3);
        acc = wmma_bf16(a, b, acc);
    }
    float bv = bias ? bias[n0 + l15] : 0.0f;
    int N = Nt << 4;
#pragma unroll
    for (int r = 0; r < 8; ++r) {
        float v = acc[r] + bv;
        if (doTanh) v = tanhf(v);
        int m = m0 + r + ((lane >> 4) << 3);
        size_t idx = (size_t)m * N + n0 + l15;
        if (Cf)  Cf[idx]  = v;
        if (Cbf) Cbf[idx] = f2bf(v);
    }
}

// ---------------------------------------------------------------------------
// persistent bidirectional encoder LSTM scan. grid=2 (blockIdx.x = direction),
// 256 threads. x@Wih^T + b precomputed in xW. Per step: WMMA h@Whh^T.
// ---------------------------------------------------------------------------
__global__ __launch_bounds__(256) void k_enc_scan(
    const float* __restrict__ xWf, const float* __restrict__ xWb,
    const unsigned short* __restrict__ WhhF, const unsigned short* __restrict__ WhhB,
    float* __restrict__ enc_hs, unsigned short* __restrict__ enc_hs_bf,
    float* __restrict__ h0, int S)
{
    const int H = 160, B = 8;
    int dir = blockIdx.x;
    const float* xW = dir ? xWb : xWf;
    const unsigned short* Whh = dir ? WhhB : WhhF;
    __shared__ unsigned short sh_hb[16 * 160];   // h (bf16, rows 8..15 zero)
    __shared__ float sh_c[8 * 160];
    __shared__ float sh_g[8 * 640];
    int tid = threadIdx.x;
    for (int i = tid; i < 16 * 160; i += 256) sh_hb[i] = 0;
    for (int i = tid; i < 8 * 160; i += 256)  sh_c[i] = 0.0f;
    __syncthreads();
    int lane = tid & 31, wave = tid >> 5;
    int l15 = lane & 15, kh = (lane >> 4) << 3;
    for (int t = 0; t < S; ++t) {
        int sidx = dir ? (S - 1 - t) : t;
        // gates(16x640) = h(16x160) @ Whh^T   (5 tiles per wave, 5 k-steps)
        for (int tt = wave; tt < 40; tt += 8) {
            int n0 = tt << 4;
            v8f acc = {0, 0, 0, 0, 0, 0, 0, 0};
            const unsigned short* pa = sh_hb + l15 * 160 + kh;
            const unsigned short* pb = Whh + (size_t)(n0 + l15) * 160 + kh;
#pragma unroll
            for (int kk = 0; kk < 5; ++kk) {
                Frag a, b;
                load_frag(pa + kk * 32, a);
                load_frag(pb + kk * 32, b);
                acc = wmma_bf16(a, b, acc);
            }
            if (lane < 16) {
                int n = n0 + l15;
#pragma unroll
                for (int r = 0; r < 8; ++r)
                    sh_g[r * 640 + n] = acc[r] + xW[((size_t)r * S + sidx) * 640 + n];
            }
        }
        __syncthreads();
        // LSTM cell
        for (int i = tid; i < B * H; i += 256) {
            int b = i / H, u = i % H;
            float gi = sh_g[b * 640 + u],       gf = sh_g[b * 640 + 160 + u];
            float gg = sh_g[b * 640 + 320 + u], go = sh_g[b * 640 + 480 + u];
            float c = sigm(gf) * sh_c[i] + sigm(gi) * tanhf(gg);
            float h = sigm(go) * tanhf(c);
            sh_c[i] = c;
            sh_hb[b * 160 + u] = f2bf(h);
            size_t o = ((size_t)b * S + sidx) * 320 + dir * 160 + u;
            enc_hs[o] = h;
            enc_hs_bf[o] = f2bf(h);
            if (t == S - 1) h0[b * 320 + dir * 160 + u] = h;
        }
        __syncthreads();
    }
}

// ---------------------------------------------------------------------------
// decoder step, part A: LSTM cell (WMMA) + temporal enc-attention +
// intra-decoder attention + concat build.  1 block, 1024 threads.
// ---------------------------------------------------------------------------
__global__ __launch_bounds__(1024) void k_dec_a(
    const unsigned short* __restrict__ decWhh,   // 1280 x 320 bf16
    const float* __restrict__ dec_xW,            // (B*T) x 1280 (x@Wih^T + b)
    const float* __restrict__ enc_keys,          // (B*S) x 320
    const float* __restrict__ enc_hs,            // (B*S) x 320
    const float* __restrict__ dec_attn_proj,     // 320 x 320
    float* __restrict__ ws_h, float* __restrict__ ws_c,
    float* __restrict__ ws_cum, float* __restrict__ ws_attn,
    float* __restrict__ dec_hs,                  // (B*T) x 320
    float* __restrict__ concat_f,                // 8 x 960
    unsigned short* __restrict__ concat_bf,      // 16 x 960 (rows 8..15 zero)
    int t, int S, int T)
{
    const int B = 8, H2 = 320;
    __shared__ float sh_h[8 * 320];
    __shared__ unsigned short sh_hb[16 * 320];
    __shared__ float sh_dyn[8 * 1280];  // gates | escores(6400)+q(2560)+dsc(1024)
    __shared__ float wred[32];
    __shared__ float bred[8];
    int tid = threadIdx.x, lane = tid & 31, wave = tid >> 5;
    int l15 = lane & 15, kh = (lane >> 4) << 3;

    // P0: load h_prev
    for (int i = tid; i < 16 * 320; i += 1024) sh_hb[i] = 0;
    __syncthreads();
    for (int i = tid; i < 8 * 320; i += 1024) {
        float h = ws_h[i]; sh_h[i] = h; sh_hb[i] = f2bf(h);
    }
    __syncthreads();

    // P1: gates(16x1280) = h @ decWhh^T  (80 tiles over 32 waves, 10 k-steps)
    float* gates = sh_dyn;
    for (int tt = wave; tt < 80; tt += 32) {
        int n0 = tt << 4;
        v8f acc = {0, 0, 0, 0, 0, 0, 0, 0};
        const unsigned short* pa = sh_hb + l15 * 320 + kh;
        const unsigned short* pb = decWhh + (size_t)(n0 + l15) * 320 + kh;
#pragma unroll
        for (int kk = 0; kk < 10; ++kk) {
            Frag a, b;
            load_frag(pa + kk * 32, a);
            load_frag(pb + kk * 32, b);
            acc = wmma_bf16(a, b, acc);
        }
        if (lane < 16) {
            int n = n0 + l15;
#pragma unroll
            for (int r = 0; r < 8; ++r)
                gates[r * 1280 + n] = acc[r] + dec_xW[((size_t)r * T + t) * 1280 + n];
        }
    }
    __syncthreads();

    // P2: LSTM cell; write h everywhere it is needed
    for (int i = tid; i < B * H2; i += 1024) {
        int b = i / H2, u = i % H2;
        float gi = gates[b * 1280 + u],        gf = gates[b * 1280 + 320 + u];
        float gg = gates[b * 1280 + 640 + u],  go = gates[b * 1280 + 960 + u];
        float c = sigm(gf) * ws_c[i] + sigm(gi) * tanhf(gg);
        float h = sigm(go) * tanhf(c);
        ws_c[i] = c; ws_h[i] = h;
        sh_h[i] = h; sh_hb[b * 320 + u] = f2bf(h);
        dec_hs[((size_t)b * T + t) * H2 + u] = h;
        concat_f[b * 960 + u] = h;
        concat_bf[b * 960 + u] = f2bf(h);
    }
    __syncthreads();

    // P3: escores[b,s] = h . enc_keys
    float* sh_e = sh_dyn;                 // 8*800
    float* sh_q = sh_dyn + 6400;          // 8*320
    float* sh_d = sh_dyn + 6400 + 2560;   // 8*128
    for (int p = tid; p < B * S; p += 1024) {
        int b = p / S, s = p % S;
        const float* k = enc_keys + (size_t)(b * S + s) * H2;
        const float* h = sh_h + b * H2;
        float acc = 0.0f;
        for (int e = 0; e < H2; e += 4)
            acc += h[e] * k[e] + h[e + 1] * k[e + 1] + h[e + 2] * k[e + 2] + h[e + 3] * k[e + 3];
        sh_e[p] = acc;
    }
    __syncthreads();

    // P4: temporal attention (exp(e)/cum) + cumulative update + softmax over s
    for (int p = tid; p < B * S; p += 1024) {
        float e = sh_e[p];
        float cum = ws_cum[p];
        float tmp = (t == 0) ? e : __expf(e) / cum;
        ws_cum[p] = cum + e;
        sh_e[p] = tmp;
    }
    __syncthreads();
    int b = tid >> 7, li = tid & 127;
    float pm = -1e30f;
    for (int s = li; s < S; s += 128) pm = fmaxf(pm, sh_e[b * S + s]);
    segreduce(pm, 1, wred, bred, tid);
    float bmax = bred[b];
    float ps = 0.0f;
    for (int s = li; s < S; s += 128) ps += __expf(sh_e[b * S + s] - bmax);
    segreduce(ps, 0, wred, bred, tid);
    float bsum = bred[b];
    for (int s = li; s < S; s += 128) {
        float a = __expf(sh_e[b * S + s] - bmax) / bsum;
        sh_e[b * S + s] = a;
        ws_attn[b * S + s] = a;
    }
    __syncthreads();

    // P5: enc_ctx = attn . enc_hs  -> concat[:,320:640]
    for (int i = tid; i < B * H2; i += 1024) {
        int bb = i / H2, d = i % H2;
        const float* hs = enc_hs + (size_t)bb * S * H2 + d;
        const float* at = sh_e + bb * S;
        float acc = 0.0f;
        for (int s = 0; s < S; ++s) acc += at[s] * hs[(size_t)s * H2];
        concat_f[bb * 960 + 320 + d] = acc;
        concat_bf[bb * 960 + 320 + d] = f2bf(acc);
    }
    // P6: q = h @ dec_attn_proj
    for (int i = tid; i < B * H2; i += 1024) {
        int bb = i / H2, d = i % H2;
        const float* h = sh_h + bb * H2;
        float acc = 0.0f;
        for (int e = 0; e < H2; ++e) acc += h[e] * dec_attn_proj[(size_t)e * H2 + d];
        sh_q[bb * H2 + d] = acc;
    }
    __syncthreads();

    // P7/P8: intra-decoder attention (tau < t) -> concat[:,640:960]
    if (t > 0) {
        for (int p = tid; p < B * 128; p += 1024) {
            int bb = p >> 7, tau = p & 127;
            float acc = -1e30f;
            if (tau < t) {
                const float* dh = dec_hs + ((size_t)bb * T + tau) * H2;
                const float* q = sh_q + bb * H2;
                acc = 0.0f;
                for (int e = 0; e < H2; e += 4)
                    acc += q[e] * dh[e] + q[e + 1] * dh[e + 1] + q[e + 2] * dh[e + 2] + q[e + 3] * dh[e + 3];
            }
            sh_d[p] = acc;
        }
        __syncthreads();
        float v = sh_d[b * 128 + li];
        segreduce(v, 1, wred, bred, tid);
        float dm = bred[b];
        float ex = __expf(v - dm);
        segreduce(ex, 0, wred, bred, tid);
        float ds = bred[b];
        sh_d[b * 128 + li] = ex / ds;
        __syncthreads();
        for (int i = tid; i < B * H2; i += 1024) {
            int bb = i / H2, d = i % H2;
            float acc = 0.0f;
            for (int tau = 0; tau < t; ++tau)
                acc += sh_d[bb * 128 + tau] * dec_hs[((size_t)bb * T + tau) * H2 + d];
            concat_f[bb * 960 + 640 + d] = acc;
            concat_bf[bb * 960 + 640 + d] = f2bf(acc);
        }
    } else {
        for (int i = tid; i < B * H2; i += 1024) {
            int bb = i / H2, d = i % H2;
            concat_f[bb * 960 + 640 + d] = 0.0f;
            concat_bf[bb * 960 + 640 + d] = 0;
        }
    }
}

// ---------------------------------------------------------------------------
// decoder step, part B: vocab logits split-softmax partials.
// one wave per 16-wide vocab tile: logits = concat(16x960) @ out_proj^T + bias
// then in-wave max / sumexp per batch row via 16-lane shuffles.
// ---------------------------------------------------------------------------
__global__ __launch_bounds__(256) void k_vocab_partials(
    const unsigned short* __restrict__ concat_bf,  // 16 x 960
    const unsigned short* __restrict__ out_proj,   // V x 960 bf16
    const float* __restrict__ out_bias,
    float* __restrict__ pmax, float* __restrict__ psum, int Ntiles)
{
    int wid = blockIdx.x * 8 + (threadIdx.x >> 5);
    if (wid >= Ntiles) return;
    int lane = threadIdx.x & 31, l15 = lane & 15, kh = (lane >> 4) << 3;
    int n0 = wid << 4;
    const unsigned short* Arow = concat_bf + l15 * 960 + kh;
    const unsigned short* Wrow = out_proj + (size_t)(n0 + l15) * 960 + kh;
    v8f acc = {0, 0, 0, 0, 0, 0, 0, 0};
#pragma unroll 5
    for (int kk = 0; kk < 30; ++kk) {
        Frag a, b;
        load_frag(Arow + kk * 32, a);
        load_frag(Wrow + kk * 32, b);
        if (kk + 1 < 30) __builtin_prefetch((const void*)(Wrow + kk * 32 + 32), 0, 3);
        acc = wmma_bf16(a, b, acc);
    }
    float bv = out_bias[n0 + l15];
#pragma unroll
    for (int r = 0; r < 8; ++r) {            // rows 0..7 = batch (lanes 0..15)
        float v = acc[r] + bv;
        float mx = v;
#pragma unroll
        for (int off = 8; off >= 1; off >>= 1) mx = fmaxf(mx, __shfl_xor(mx, off));
        float ex = __expf(v - mx), sm = ex;
#pragma unroll
        for (int off = 8; off >= 1; off >>= 1) sm += __shfl_xor(sm, off);
        if (lane == 0) { pmax[r * Ntiles + wid] = mx; psum[r * Ntiles + wid] = sm; }
    }
}

// ---------------------------------------------------------------------------
// decoder step, part C: merge split-softmax, pointer-generator mix, loss.
// ---------------------------------------------------------------------------
__global__ __launch_bounds__(1024) void k_dec_c(
    const float* __restrict__ pmax, const float* __restrict__ psum,
    const float* __restrict__ concat_f, const unsigned short* __restrict__ out_proj,
    const float* __restrict__ out_bias, const float* __restrict__ ws_attn,
    const int* __restrict__ input_ids, const int* __restrict__ labels,
    const int* __restrict__ input_lengths,
    const float* __restrict__ switch_W, const float* __restrict__ switch_b,
    float* __restrict__ ws_loss, int* __restrict__ ws_ntok, float* __restrict__ d_out,
    int t, int S, int T, int V, int Ntiles)
{
    __shared__ float wred[32];
    __shared__ float s_gmax[8], s_gsum[8], s_lt[8], s_cp[8], s_pc[8], s_nll[8];
    int tid = threadIdx.x;
    int b = tid >> 7, li = tid & 127;

    int tok = labels[b * T + t];
    int tokc = tok < 0 ? 0 : (tok >= V ? V - 1 : tok);

    float m = -1e30f;
    for (int i = li; i < Ntiles; i += 128) m = fmaxf(m, pmax[b * Ntiles + i]);
    segreduce(m, 1, wred, s_gmax, tid);
    float gm = s_gmax[b];

    float s0 = 0.0f;
    for (int i = li; i < Ntiles; i += 128)
        s0 += psum[b * Ntiles + i] * __expf(pmax[b * Ntiles + i] - gm);
    segreduce(s0, 0, wred, s_gsum, tid);

    float lt = 0.0f;
    for (int j = li; j < 960; j += 128)
        lt += concat_f[b * 960 + j] * bf2f(out_proj[(size_t)tokc * 960 + j]);
    segreduce(lt, 0, wred, s_lt, tid);

    float cp = 0.0f;
    for (int s = li; s < S; s += 128)
        cp += ws_attn[b * S + s] * ((input_ids[b * S + s] == tok) ? 1.0f : 0.0f);
    segreduce(cp, 0, wred, s_cp, tid);

    float pd = 0.0f;
    for (int j = li; j < 960; j += 128)
        pd += concat_f[b * 960 + j] * switch_W[j];
    segreduce(pd, 0, wred, s_pc, tid);

    if (li == 0) {
        float gsum = s_gsum[b];
        float logit = s_lt[b] + out_bias[tokc];
        float gen_p = (tok < V) ? __expf(logit - gm) / gsum : 0.0f;
        float pcopy = sigm(s_pc[b] + switch_b[0]);
        float p = (1.0f - pcopy) * gen_p + pcopy * s_cp[b];
        s_nll[b] = (input_lengths[b] <= t) ? 0.0f : -logf(p + 1e-9f);
    }
    __syncthreads();
    if (tid == 0) {
        int cnt = 0;
        for (int bb = 0; bb < 8; ++bb) {
            ws_loss[bb] += s_nll[bb];
            if (input_lengths[bb] > t) cnt++;
        }
        *ws_ntok += cnt;
        if (t == T - 1) {
            float tot = 0.0f;
            for (int bb = 0; bb < 8; ++bb) tot += ws_loss[bb];
            d_out[0] = tot / (float)(*ws_ntok);
        }
    }
}

// ---------------------------------------------------------------------------
// host-side orchestration
// ---------------------------------------------------------------------------
extern "C" void kernel_launch(void* const* d_in, const int* in_sizes, int n_in,
                              void* d_out, int out_size, void* d_ws, size_t ws_size,
                              hipStream_t stream) {
    (void)in_sizes; (void)n_in; (void)out_size; (void)ws_size;
    const int V = 50000, E = 128, H = 160, B = 8, S = 800, T = 100;
    const int H2 = 2 * H, N6 = 6 * H;            // 320, 960
    const int NT = V / 16;                       // 3125 vocab tiles

    const int*   input_ids     = (const int*)d_in[0];
    const int*   labels        = (const int*)d_in[1];
    const int*   input_lengths = (const int*)d_in[2];
    const float* embedding     = (const float*)d_in[3];
    const float* enc_Wih_f     = (const float*)d_in[4];
    const float* enc_Whh_f     = (const float*)d_in[5];
    const float* enc_b_f       = (const float*)d_in[6];
    const float* enc_Wih_b     = (const float*)d_in[7];
    const float* enc_Whh_b     = (const float*)d_in[8];
    const float* enc_b_b       = (const float*)d_in[9];
    const float* dec_Wih       = (const float*)d_in[10];
    const float* dec_Whh       = (const float*)d_in[11];
    const float* dec_b         = (const float*)d_in[12];
    const float* enc_attn_proj = (const float*)d_in[13];
    const float* dec_attn_proj = (const float*)d_in[14];
    const float* vocab_proj    = (const float*)d_in[15];
    const float* out_bias      = (const float*)d_in[16];
    const float* switch_W      = (const float*)d_in[17];
    const float* switch_b      = (const float*)d_in[18];

    char* W = (char*)d_ws;
    size_t off = 0;
    auto alloc = [&](size_t bytes) {
        size_t o = off; off = (off + bytes + 255) & ~(size_t)255; return o;
    };
    // ---- zero-initialized region (contiguous, starts at 0) ----
    size_t o_dechs = alloc((size_t)B * T * H2 * 4);
    size_t o_c     = alloc((size_t)B * H2 * 4);
    size_t o_cum   = alloc((size_t)B * S * 4);
    size_t o_cbf   = alloc((size_t)16 * N6 * 2);      // concat bf16 (rows 8..15 stay 0)
    size_t o_loss  = alloc(8 * 4);
    size_t o_ntok  = alloc(256);
    size_t zwords  = off / 4;
    // ---- remaining scratch ----
    size_t o_h     = alloc((size_t)B * H2 * 4);
    size_t o_embbf = alloc((size_t)V * E * 2);
    size_t o_vpT   = alloc((size_t)N6 * E * 2);
    size_t o_WihF  = alloc((size_t)4 * H * E * 2);
    size_t o_WhhF  = alloc((size_t)4 * H * H * 2);
    size_t o_WihB  = alloc((size_t)4 * H * E * 2);
    size_t o_WhhB  = alloc((size_t)4 * H * H * 2);
    size_t o_dWih  = alloc((size_t)8 * H * E * 2);
    size_t o_dWhh  = alloc((size_t)8 * H * H2 * 2);
    size_t o_eap   = alloc((size_t)H2 * H2 * 2);
    size_t o_x     = alloc((size_t)B * S * E * 2);
    size_t o_dx    = alloc((size_t)B * T * E * 2);
    size_t o_xWf   = alloc((size_t)B * S * 4 * H * 4);
    size_t o_xWb   = alloc((size_t)B * S * 4 * H * 4);
    size_t o_dxW   = alloc((size_t)B * T * 8 * H * 4);
    size_t o_op    = alloc((size_t)V * N6 * 2);       // out_proj bf16, 96 MB (L2-resident)
    size_t o_ehs   = alloc((size_t)B * S * H2 * 4);
    size_t o_ehsb  = alloc((size_t)B * S * H2 * 2);
    size_t o_ekeys = alloc((size_t)B * S * H2 * 4);
    size_t o_attn  = alloc((size_t)B * S * 4);
    size_t o_cf    = alloc((size_t)B * N6 * 4);
    size_t o_pmax  = alloc((size_t)8 * NT * 4);
    size_t o_psum  = alloc((size_t)8 * NT * 4);

    #define F32(o)  ((float*)(W + (o)))
    #define BF(o)   ((unsigned short*)(W + (o)))

    auto blks = [](int n, int bs) { return (n + bs - 1) / bs; };

    // 1. init
    k_zero_u32<<<blks((int)zwords, 256), 256, 0, stream>>>((unsigned*)W, (int)zwords);

    // 2. weight conversions to bf16
    k_cvt_bf16<<<blks(V * E, 256), 256, 0, stream>>>(embedding, BF(o_embbf), V * E);
    k_cvt_bf16<<<blks(4 * H * E, 256), 256, 0, stream>>>(enc_Wih_f, BF(o_WihF), 4 * H * E);
    k_cvt_bf16<<<blks(4 * H * H, 256), 256, 0, stream>>>(enc_Whh_f, BF(o_WhhF), 4 * H * H);
    k_cvt_bf16<<<blks(4 * H * E, 256), 256, 0, stream>>>(enc_Wih_b, BF(o_WihB), 4 * H * E);
    k_cvt_bf16<<<blks(4 * H * H, 256), 256, 0, stream>>>(enc_Whh_b, BF(o_WhhB), 4 * H * H);
    k_cvt_bf16<<<blks(8 * H * E, 256), 256, 0, stream>>>(dec_Wih, BF(o_dWih), 8 * H * E);
    k_cvt_bf16<<<blks(8 * H * H2, 256), 256, 0, stream>>>(dec_Whh, BF(o_dWhh), 8 * H * H2);
    k_cvt_bf16<<<blks(H2 * H2, 256), 256, 0, stream>>>(enc_attn_proj, BF(o_eap), H2 * H2);
    k_transpose_bf16<<<blks(E * N6, 256), 256, 0, stream>>>(vocab_proj, BF(o_vpT), E, N6);

    // 3. embedding gathers (bf16)
    k_gather_enc<<<blks(B * S * E, 256), 256, 0, stream>>>(input_ids, embedding, BF(o_x), B * S, E, V);
    k_gather_dec<<<blks(B * T * E, 256), 256, 0, stream>>>(labels, embedding, BF(o_dx), B, T, E, V);

    // 4. batched WMMA GEMMs
    // xWf/xWb: (B*S,E)@(4H,E)^T + b     dec_xW: (B*T,E)@(8H,E)^T + b
    k_wmma_gemm<<<blks((B * S / 16) * (4 * H / 16), 8), 256, 0, stream>>>(
        BF(o_x), BF(o_WihF), F32(o_xWf), nullptr, enc_b_f, B * S / 16, 4 * H / 16, E, 0);
    k_wmma_gemm<<<blks((B * S / 16) * (4 * H / 16), 8), 256, 0, stream>>>(
        BF(o_x), BF(o_WihB), F32(o_xWb), nullptr, enc_b_b, B * S / 16, 4 * H / 16, E, 0);
    k_wmma_gemm<<<blks((B * T / 16) * (8 * H / 16), 8), 256, 0, stream>>>(
        BF(o_dx), BF(o_dWih), F32(o_dxW), nullptr, dec_b, B * T / 16, 8 * H / 16, E, 0);
    // out_proj = tanh(emb @ vocab_proj): (V,E)@(N6,E)^T -> bf16 (96 MB)
    k_wmma_gemm<<<blks((V / 16) * (N6 / 16), 8), 256, 0, stream>>>(
        BF(o_embbf), BF(o_vpT), nullptr, BF(o_op), nullptr, V / 16, N6 / 16, E, 1);

    // 5. persistent bidirectional encoder scan (2 WGPs in parallel)
    k_enc_scan<<<2, 256, 0, stream>>>(F32(o_xWf), F32(o_xWb), BF(o_WhhF), BF(o_WhhB),
                                      F32(o_ehs), BF(o_ehsb), F32(o_h), S);

    // 6. enc_keys = enc_hs @ enc_attn_proj^T
    k_wmma_gemm<<<blks((B * S / 16) * (H2 / 16), 8), 256, 0, stream>>>(
        BF(o_ehsb), BF(o_eap), F32(o_ekeys), nullptr, nullptr, B * S / 16, H2 / 16, H2, 0);

    // 7. decoder: 100 sequential steps, 3 kernels each
    for (int t = 0; t < T; ++t) {
        k_dec_a<<<1, 1024, 0, stream>>>(
            BF(o_dWhh), F32(o_dxW), F32(o_ekeys), F32(o_ehs), dec_attn_proj,
            F32(o_h), F32(o_c), F32(o_cum), F32(o_attn), F32(o_dechs),
            F32(o_cf), BF(o_cbf), t, S, T);
        k_vocab_partials<<<blks(NT, 8), 256, 0, stream>>>(
            BF(o_cbf), BF(o_op), out_bias, F32(o_pmax), F32(o_psum), NT);
        k_dec_c<<<1, 1024, 0, stream>>>(
            F32(o_pmax), F32(o_psum), F32(o_cf), BF(o_op), out_bias, F32(o_attn),
            input_ids, labels, input_lengths, switch_W, switch_b,
            F32(o_loss), (int*)(W + o_ntok), (float*)d_out, t, S, T, V, NT);
    }
    #undef F32
    #undef BF
}
